// BasicYATBlock_19791209300356
// MI455X (gfx1250) — matrix operand order repaired and checked by reading, not compiled, not run on hardware
//
#include <hip/hip_runtime.h>

typedef __attribute__((ext_vector_type(16))) __bf16 v16bf;
typedef __attribute__((ext_vector_type(8)))  float  v8f;

#define YAT_EPS 0.007f

// --------------------------------------------------------------------------
// CDNA5 async global->LDS path (ASYNCcnt-tracked, bypasses VGPRs).
// Builtin signature (from compiler diagnostics):
//   void __builtin_amdgcn_global_load_async_to_lds_b128(
//       int4* addrspace(1) src, int4* addrspace(3) dst, imm offset, imm cpol)
// --------------------------------------------------------------------------
#if __has_builtin(__builtin_amdgcn_global_load_async_to_lds_b128)
#define USE_ASYNC_LDS 1
#else
#define USE_ASYNC_LDS 0
#endif

typedef int i4v __attribute__((__vector_size__(4 * sizeof(int))));
#define GLB_AS(p) ((__attribute__((address_space(1))) i4v*)(p))
#define LDS_AS(p) ((__attribute__((address_space(3))) i4v*)(p))

__device__ __forceinline__ void wait_async0() {
#if __has_builtin(__builtin_amdgcn_s_wait_asynccnt)
  __builtin_amdgcn_s_wait_asynccnt(0);
#else
  asm volatile("s_wait_asynccnt 0x0" ::: "memory");
#endif
}

// ---------------------------------------------------------------------------
// Weight packing: OIHW f32 -> bf16 Bp[kb][n][kin], K ordered (kh,kw,ci) so a
// 32-wide K slice maps to 32 contiguous channels of one tap in NHWC activations.
// ---------------------------------------------------------------------------
__global__ __launch_bounds__(256) void pack_w(const float* __restrict__ w,
                                              __bf16* __restrict__ Bp,
                                              int Cin, int KHW) {
  int tid = blockIdx.x * 256 + threadIdx.x;   // == output offset
  int kin = tid & 31;
  int n   = (tid >> 5) & 255;
  int kb  = tid >> 13;
  int k   = kb * 32 + kin;                    // k = tap*Cin + ci
  int tap = k / Cin;
  int ci  = k - tap * Cin;
  int kh  = tap / 3;
  int kw  = tap - kh * 3;
  float v = w[(((size_t)n * Cin + ci) * KHW + kh) * KHW + kw];
  Bp[tid] = (__bf16)v;
}

// Per-filter squared norms of w_yat (contiguous 1152 floats per output channel)
__global__ void wsq_kernel(const float* __restrict__ w, float* __restrict__ wsq) {
  int n = threadIdx.x;                        // 256 threads
  const float* p = w + (size_t)n * 1152;
  float s = 0.f;
  for (int i = 0; i < 1152; ++i) { float v = p[i]; s += v * v; }
  wsq[n] = s;
}

// x: NCHW f32 [32][128][56][56] -> NHWC bf16 [32][56][56][128]
__global__ __launch_bounds__(256) void nchw_to_nhwc_bf16(const float* __restrict__ x,
                                                         __bf16* __restrict__ xb) {
  int tid = blockIdx.x * 256 + threadIdx.x;   // 12,845,056 total
  int w = tid % 56;
  int h = (tid / 56) % 56;
  int c = (tid / 3136) % 128;
  int n = tid / 401408;
  xb[(((size_t)n * 56 + h) * 56 + w) * 128 + c] = (__bf16)x[tid];
}

// patch_sq: sum of x^2 over each 3x3 (stride 2, pad 1) patch, all 128 channels
__global__ __launch_bounds__(256) void psq_kernel(const __bf16* __restrict__ xb,
                                                  float* __restrict__ psq) {
  int m = blockIdx.x * 256 + threadIdx.x;     // 25088 outputs
  int nimg = m / 784;
  int p    = m - nimg * 784;
  int oh = p / 28, ow = p - (p / 28) * 28;
  float s = 0.f;
  for (int kh = 0; kh < 3; ++kh) {
    int ih = oh * 2 + kh - 1;
    if (ih < 0 || ih >= 56) continue;
    for (int kw = 0; kw < 3; ++kw) {
      int iw = ow * 2 + kw - 1;
      if (iw < 0 || iw >= 56) continue;
      const __bf16* q = xb + (((size_t)nimg * 56 + ih) * 56 + iw) * 128;
#pragma unroll 8
      for (int c = 0; c < 128; ++c) { float v = (float)q[c]; s += v * v; }
    }
  }
  psq[m] = s;
}

// ---------------------------------------------------------------------------
// Implicit-GEMM conv via WMMA bf16->f32 with double-buffered async LDS stage.
// Block tile 128(M) x 64(N), 256 threads = 8 wave32 in 4x2; wave tile 32x32
// = 2x2 accumulators of 16x16. K stepped 32/iter; copy(k+1) overlaps mma(k).
// ---------------------------------------------------------------------------
template <int CIN, int HIN, int WIN, int STRIDE, int PAD, int KTOT,
          bool YAT_EPI, bool ADD_ID>
__global__ __launch_bounds__(256) void conv_gemm(
    const __bf16* __restrict__ A,      // NHWC [32][HIN][WIN][CIN]
    const __bf16* __restrict__ Bp,     // [KTOT/32][256][32]
    const __bf16* __restrict__ xid,    // NHWC [32][56][56][128] (shortcut src)
    const __bf16* __restrict__ B3p,    // [4][256][32] (shortcut weights)
    const float*  __restrict__ psq,
    const float*  __restrict__ wsq,
    const float*  __restrict__ alpha_p,
    __bf16* __restrict__ ybuf,         // YAT output, [M][256] bf16 (== NHWC)
    float*  __restrict__ outp)         // final output, NCHW f32
{
  constexpr int OH = 28, OW = 28;
  constexpr int KB_MAIN = KTOT / 32;
  constexpr int KB_ALL  = KB_MAIN + (ADD_ID ? 4 : 0);

  __shared__ __align__(64) __bf16 ldsA[2][128 * 32];  // [buf][m_local][k_local]
  __shared__ __align__(64) __bf16 ldsB[2][64 * 32];   // [buf][n_local][k_local]

  const int t    = threadIdx.x;
  const int lane = t & 31;
  const int wave = t >> 5;
  const int wm   = wave & 3;          // 4 waves along M
  const int wn   = wave >> 2;         // 2 waves along N
  const int l15  = lane & 15;
  const int ksel = lane >> 4;         // K-half select per ISA fragment layout

  const int Mtile = blockIdx.x;       // 196
  const int Ntile = blockIdx.y;       // 4

  // A-tile fill coordinates: 2 threads per m-row, 16 bf16 each
  const int m_local = t >> 1;
  const int hsel    = t & 1;
  const int m    = Mtile * 128 + m_local;
  const int nimg = m / (OH * OW);
  const int prem = m - nimg * (OH * OW);
  const int oh   = prem / OW;
  const int ow   = prem - oh * OW;

  const int nl   = t >> 2;            // B-tile fill: 4 threads per n-row
  const int koff = (t & 3) * 8;

  const v8f vzero = {0.f, 0.f, 0.f, 0.f, 0.f, 0.f, 0.f, 0.f};
  v8f acc[2][2];
#pragma unroll
  for (int i = 0; i < 2; ++i)
#pragma unroll
    for (int j = 0; j < 2; ++j) acc[i][j] = vzero;

  union Frag { v16bf v; uint4 q[2]; };

  // Issue the copy of K-block kbi into LDS buffer `buf`.
  auto stage = [&](int kbi, int buf) {
    // --- A operand (implicit im2col; one tap, 16 contiguous channels) ---
    const __bf16* srcA;
    bool valid;
    if (ADD_ID && kbi >= KB_MAIN) {   // fused 1x1 stride-2 shortcut stages
      int kg = (kbi - KB_MAIN) * 32 + hsel * 16;  // == ci0, always in bounds
      srcA  = xid + (((size_t)nimg * 56 + oh * 2) * 56 + ow * 2) * 128 + kg;
      valid = true;
    } else {
      int kg  = kbi * 32 + hsel * 16;
      int tap = kg / CIN;
      int ci0 = kg - tap * CIN;
      int kh = tap / 3, kw = tap - (tap / 3) * 3;
      int ih = oh * STRIDE + kh - PAD;
      int iw = ow * STRIDE + kw - PAD;
      valid = (ih >= 0 && ih < HIN && iw >= 0 && iw < WIN);
      srcA  = A + (((size_t)nimg * HIN + (valid ? ih : 0)) * WIN +
                   (valid ? iw : 0)) * CIN + ci0;
    }
    __bf16* dA = &ldsA[buf][m_local * 32 + hsel * 16];

    // --- B operand (packed tile: straight contiguous copy) ---
    const __bf16* srcB =
        (ADD_ID && kbi >= KB_MAIN)
            ? B3p + (((size_t)(kbi - KB_MAIN) * 256) + Ntile * 64 + nl) * 32 + koff
            : Bp  + (((size_t)kbi * 256) + Ntile * 64 + nl) * 32 + koff;
    __bf16* dB = &ldsB[buf][nl * 32 + koff];

#if USE_ASYNC_LDS
    if (valid) {
      __builtin_amdgcn_global_load_async_to_lds_b128(GLB_AS(srcA), LDS_AS(dA), 0, 0);
      __builtin_amdgcn_global_load_async_to_lds_b128(GLB_AS(srcA), LDS_AS(dA), 16, 0);
    } else {
      uint4 z = {0, 0, 0, 0};
      uint4* d = (uint4*)dA;
      d[0] = z; d[1] = z;               // padded rows: zero via DS (dscnt-fenced)
    }
    __builtin_amdgcn_global_load_async_to_lds_b128(GLB_AS(srcB), LDS_AS(dB), 0, 0);
#else
    uint4 lo = {0, 0, 0, 0}, hi = {0, 0, 0, 0};
    if (valid) {
      const uint4* s = (const uint4*)srcA;
      lo = s[0]; hi = s[1];
    }
    uint4* d = (uint4*)dA;
    d[0] = lo; d[1] = hi;
    *(uint4*)dB = *(const uint4*)srcB;
#endif
  };

  // One 16x16x32 MMA step over the staged LDS tiles (4 wmma per call)
  auto mma_step = [&](int buf) {
    v16bf afrag[2], bfrag[2];
#pragma unroll
    for (int i = 0; i < 2; ++i) {
      // A 16x32 bf16: lanes 0-15 K=0..7 & 16..23, lanes 16-31 K=8..15 & 24..31
      const __bf16* ap = &ldsA[buf][(wm * 32 + i * 16 + l15) * 32 + ksel * 8];
      Frag u;
      u.q[0] = *(const uint4*)ap;
      u.q[1] = *(const uint4*)(ap + 16);
      afrag[i] = u.v;
    }
#pragma unroll
    for (int j = 0; j < 2; ++j) {
      // B 32x16 bf16: lane = N, lanes 0-15 K=0..7 & 16..23, lanes 16-31 K=8..15 & 24..31
      const __bf16* bp = &ldsB[buf][(wn * 32 + j * 16 + l15) * 32 + ksel * 8];
      Frag u;
      u.q[0] = *(const uint4*)bp;
      u.q[1] = *(const uint4*)(bp + 16);
      bfrag[j] = u.v;
    }
#pragma unroll
    for (int i = 0; i < 2; ++i)
#pragma unroll
      for (int j = 0; j < 2; ++j)
        acc[i][j] = __builtin_amdgcn_wmma_f32_16x16x32_bf16(
            false, afrag[i], false, bfrag[j], (short)0, acc[i][j], false, false);
  };

  // Software pipeline: wait(tile k) -> barrier -> issue copy(tile k+1) -> mma(k).
  stage(0, 0);
#pragma unroll 1
  for (int kb = 0; kb < KB_ALL; ++kb) {
    const int buf = kb & 1;
#if USE_ASYNC_LDS
    wait_async0();                    // this wave's copies for tile kb done
#endif
    __syncthreads();                  // all waves' copies visible; prior reads done
    if (kb + 1 < KB_ALL) stage(kb + 1, buf ^ 1);
    mma_step(buf);
  }

  // Epilogue. C/D layout: lane l, VGPR r -> M = r + 8*(l>>4), N = l&15.
  float scale = 1.f;
  if constexpr (YAT_EPI) {
    float a = alpha_p[0];
    scale = __powf(16.0f / __logf(257.0f), a);  // (sqrt(256)/log1p(256))^alpha
  }
#pragma unroll
  for (int i = 0; i < 2; ++i)
#pragma unroll
    for (int j = 0; j < 2; ++j)
#pragma unroll
      for (int r = 0; r < 8; ++r) {
        int mr = Mtile * 128 + wm * 32 + i * 16 + ksel * 8 + r;
        int nc = Ntile * 64 + wn * 32 + j * 16 + l15;
        float v = acc[i][j][r];
        if constexpr (YAT_EPI) {
          float dist = psq[mr] + wsq[nc] - 2.0f * v;
          float y = (v * v) / (dist + YAT_EPS) * scale;
          ybuf[(size_t)mr * 256 + nc] = (__bf16)y;
        } else {
          int ni = mr / 784;
          int pp = mr - ni * 784;
          outp[((size_t)ni * 256 + nc) * 784 + pp] = v;
        }
      }
}

// ---------------------------------------------------------------------------
extern "C" void kernel_launch(void* const* d_in, const int* in_sizes, int n_in,
                              void* d_out, int out_size, void* d_ws, size_t ws_size,
                              hipStream_t stream) {
  (void)in_sizes; (void)n_in; (void)out_size; (void)ws_size;
  const float* x       = (const float*)d_in[0];  // [32,128,56,56]
  const float* w_yat   = (const float*)d_in[1];  // [256,128,3,3]
  const float* alpha   = (const float*)d_in[2];  // [1]
  const float* w_lin   = (const float*)d_in[3];  // [256,256,3,3]
  const float* w_short = (const float*)d_in[4];  // [256,128,1,1]
  float* out = (float*)d_out;                    // [32,256,28,28]

  char* ws = (char*)d_ws;
  size_t off = 0;
  auto carve = [&](size_t bytes) -> void* {
    void* p = ws + off;
    off = (off + bytes + 255) & ~(size_t)255;
    return p;
  };
  __bf16* xb   = (__bf16*)carve(12845056ull * 2);      // x NHWC bf16
  __bf16* ybuf = (__bf16*)carve(25088ull * 256 * 2);   // YAT out NHWC bf16
  __bf16* B1p  = (__bf16*)carve(1152ull * 256 * 2);    // packed w_yat
  __bf16* B2p  = (__bf16*)carve(2304ull * 256 * 2);    // packed w_lin
  __bf16* B3p  = (__bf16*)carve(128ull * 256 * 2);     // packed w_short
  float*  wsq  = (float*)carve(256 * 4);
  float*  psq  = (float*)carve(25088ull * 4);

  pack_w<<<1152, 256, 0, stream>>>(w_yat, B1p, 128, 3);
  pack_w<<<2304, 256, 0, stream>>>(w_lin, B2p, 256, 3);
  pack_w<<<128, 256, 0, stream>>>(w_short, B3p, 128, 1);
  wsq_kernel<<<1, 256, 0, stream>>>(w_yat, wsq);
  nchw_to_nhwc_bf16<<<50176, 256, 0, stream>>>(x, xb);
  psq_kernel<<<98, 256, 0, stream>>>(xb, psq);

  dim3 grid(196, 4);  // 25088/128 M-tiles x 256/64 N-tiles
  // YAT conv: 3x3 s2 p1, K=1152, fused dist/eps epilogue -> ybuf (bf16 NHWC)
  conv_gemm<128, 56, 56, 2, 1, 1152, true, false><<<grid, 256, 0, stream>>>(
      xb, B1p, nullptr, nullptr, psq, wsq, alpha, ybuf, nullptr);
  // Final conv: 3x3 s1 p1, K=2304, + fused 1x1 s2 shortcut -> out (f32 NCHW)
  conv_gemm<256, 28, 28, 1, 1, 2304, false, true><<<grid, 256, 0, stream>>>(
      ybuf, B2p, xb, B3p, nullptr, nullptr, nullptr, nullptr, out);
}
